// MambaBlock_16140487098927
// MI455X (gfx1250) — matrix operand order repaired and checked by reading, not compiled, not run on hardware
//
#include <hip/hip_runtime.h>

// Problem constants (from reference)
#define D_MODEL 256
#define D_INNER 512
#define D_STATE 16
#define D_CONV  4
#define DT_RANK 16
#define BATCH   4
#define SEQ     4096
#define M_ROWS  (BATCH * SEQ)      // 16384 rows for all GEMMs
#define NCHUNK  64
#define CLEN    (SEQ / NCHUNK)     // 64

typedef __attribute__((ext_vector_type(16))) __bf16        v16bf;
typedef __attribute__((ext_vector_type(8)))  float         v8f;
typedef __attribute__((ext_vector_type(4)))  unsigned int  u32x4;

__device__ __forceinline__ unsigned short f2bf(float f) {
  unsigned int u = __float_as_uint(f);
  u += 0x7FFFu + ((u >> 16) & 1u);           // round-to-nearest-even
  return (unsigned short)(u >> 16);
}

__device__ __forceinline__ float fast_sigmoid(float x) {
  return 1.0f / (1.0f + __expf(-x));
}

// softplus(bias + <xr[0:16], wdt>)
__device__ __forceinline__ float compute_dt(const float* __restrict__ xr,
                                            const float* wdt, float bias) {
  float acc = bias;
#pragma unroll
  for (int r = 0; r < DT_RANK; ++r) acc += xr[r] * wdt[r];
  return (acc > 20.0f) ? acc : __logf(1.0f + __expf(acc));
}

// ---------------- f32 -> bf16 convert ----------------
__global__ __launch_bounds__(256) void k_f32_to_bf16(
    const float* __restrict__ in, unsigned short* __restrict__ out, int n) {
  int i = blockIdx.x * blockDim.x + threadIdx.x;
  if (i < n) out[i] = f2bf(in[i]);
}

// ---------------- bf16 WMMA GEMM, 4x(NSUB) register-blocked ----------------
// C[M x N] = A[M x K] * W^T,  W stored row-major (N x K).
// Block = 128 threads = 4 waves; __launch_bounds__(128,1) so the full VGPR file
// is available (4x4 blocking needs ~175 VGPRs: 128 accum + 32 A-frag + 8 B-frag).
// Strides are template constants so per-tile row offsets fold into the 24-bit
// instruction offset of global_load_b128 (single base pointer per matrix).
// Wave w computes a 64 x (16*NSUB) output block:
//   rows  [ (blockIdx.x*4+w)*64 , +64 )      (4 m-tiles)
//   cols  [ blockIdx.y*16*NSUB  , +16*NSUB ) (NSUB n-tiles)
// Per K-step (32): 8 A + 2*NSUB B b128 loads -> 4*NSUB WMMAs (1 load/WMMA @ NSUB=4).
// VGPR layouts per CDNA5 ISA 7.12.2:
//   A 16x32 bf16: lane m (0-15) holds row m, elems 0..7 -> K k0..k0+7,
//                 elems 8..15 -> K k0+16..k0+23; lanes 16-31 same rows, K +8.
//   B 32x16 bf16: lane n (0-15) holds col n, same K order as A (elems 0..7 -> k0..k0+7,
//                 elems 8..15 -> k0+16..k0+23; lanes 16-31 K +8).
//   C/D f32:      VGPR r of lane ln -> (row r + 8*(ln>=16), col ln&15).
template <int NSUB, int LDA, int LDW, int LDC>
__global__ __launch_bounds__(128, 1) void k_wmma_gemm_bf16(
    const unsigned short* __restrict__ A,
    const unsigned short* __restrict__ W,
    float* __restrict__ C, int K) {
  const int wave = threadIdx.x >> 5;
  const int lane = threadIdx.x & 31;
  const int mg   = blockIdx.x * 4 + wave;   // m-group (4 tiles = 64 rows)
  const int ng   = blockIdx.y;              // n-group (NSUB tiles)
  const int lo   = lane & 15;
  const int hi   = lane >> 4;

  const unsigned short* Abase = A + (size_t)(mg * 64 + lo) * LDA + hi * 8;
  const unsigned short* Wbase = W + (size_t)(ng * (NSUB * 16) + lo) * LDW + hi * 8;

  v8f acc[4][NSUB];
#pragma unroll
  for (int i = 0; i < 4; ++i)
#pragma unroll
    for (int j = 0; j < NSUB; ++j)
      acc[i][j] = (v8f){0.f, 0.f, 0.f, 0.f, 0.f, 0.f, 0.f, 0.f};

  union Frag { v16bf v; u32x4 q[2]; };
  for (int k0 = 0; k0 < K; k0 += 32) {
    Frag af[4];
#pragma unroll
    for (int i = 0; i < 4; ++i) {          // row offset i*16*LDA is a constant
      af[i].q[0] = *(const u32x4*)(Abase + i * 16 * LDA + k0);
      af[i].q[1] = *(const u32x4*)(Abase + i * 16 * LDA + k0 + 16);
    }
#pragma unroll
    for (int j = 0; j < NSUB; ++j) {
      Frag bf_;
      bf_.q[0] = *(const u32x4*)(Wbase + j * 16 * LDW + k0);
      bf_.q[1] = *(const u32x4*)(Wbase + j * 16 * LDW + k0 + 16);
#pragma unroll
      for (int i = 0; i < 4; ++i)
        acc[i][j] = __builtin_amdgcn_wmma_f32_16x16x32_bf16(
            false, af[i].v, false, bf_.v, (short)0, acc[i][j], false, false);
    }
  }

#pragma unroll
  for (int i = 0; i < 4; ++i) {
    float* Cp = C + (size_t)(mg * 64 + i * 16 + hi * 8) * LDC;
#pragma unroll
    for (int j = 0; j < NSUB; ++j) {
      const int ncol = ng * (NSUB * 16) + j * 16 + lo;
#pragma unroll
      for (int r = 0; r < 8; ++r) Cp[r * LDC + ncol] = acc[i][j][r];
    }
  }
}

// ---------------- causal depthwise conv1d + SiLU ----------------
// x_in = xz[:, 0:512]; writes f32 (for scan) and bf16 (for x_proj GEMM).
__global__ __launch_bounds__(256) void k_conv_silu(
    const float* __restrict__ xz, const float* __restrict__ cw,
    const float* __restrict__ cb, float* __restrict__ xcf,
    unsigned short* __restrict__ xcbf) {
  int idx = blockIdx.x * blockDim.x + threadIdx.x;   // over M_ROWS*D_INNER
  if (idx >= M_ROWS * D_INNER) return;
  int d = idx & (D_INNER - 1);
  int m = idx >> 9;
  int l = m & (SEQ - 1);
  const float* w = cw + d * D_CONV;
  float acc = cb[d];
#pragma unroll
  for (int j = 0; j < D_CONV; ++j) {
    int ls = l + j - (D_CONV - 1);
    if (ls >= 0) acc += w[j] * xz[(size_t)(m + ls - l) * (2 * D_INNER) + d];
  }
  float s = acc * fast_sigmoid(acc);
  xcf[idx]  = s;
  xcbf[idx] = f2bf(s);
}

// ---------------- chunked selective scan ----------------
// Pass 1: per (b, chunk, d) compute chunk-local (prod dA, accumulated dBu) per state.
__global__ __launch_bounds__(128, 1) void k_scan_pass1(
    const float* __restrict__ xdbl, const float* __restrict__ xcf,
    const float* __restrict__ dtw, const float* __restrict__ dtb,
    const float* __restrict__ A_log,
    float* __restrict__ Aprod, float* __restrict__ Bacc) {
  int tid   = blockIdx.x * blockDim.x + threadIdx.x;   // [0, BATCH*NCHUNK*D_INNER)
  int d     = tid & (D_INNER - 1);
  int chunk = (tid >> 9) & (NCHUNK - 1);
  int b     = tid >> 15;

  float wdt[DT_RANK], a[D_STATE], ap[D_STATE], bc[D_STATE];
#pragma unroll
  for (int r = 0; r < DT_RANK; ++r) wdt[r] = dtw[d * DT_RANK + r];
#pragma unroll
  for (int s = 0; s < D_STATE; ++s) {
    a[s]  = -__expf(A_log[d * D_STATE + s]);
    ap[s] = 1.0f;
    bc[s] = 0.0f;
  }
  float bias = dtb[d];

  size_t m0 = (size_t)b * SEQ + (size_t)chunk * CLEN;
  for (int l = 0; l < CLEN; ++l) {
    size_t m = m0 + l;
    const float* xr = xdbl + m * 48;
    __builtin_prefetch(xr + 48, 0, 0);   // warm next dt/B/C row (global_prefetch_b8)
    float dtv = compute_dt(xr, wdt, bias);
    float dtx = dtv * xcf[m * D_INNER + d];
#pragma unroll
    for (int s = 0; s < D_STATE; ++s) {
      float dA = __expf(dtv * a[s]);
      bc[s] = dA * bc[s] + dtx * xr[16 + s];
      ap[s] *= dA;
    }
  }
  size_t o = (size_t)tid * D_STATE;
#pragma unroll
  for (int s = 0; s < D_STATE; ++s) { Aprod[o + s] = ap[s]; Bacc[o + s] = bc[s]; }
}

// Pass 2: serial combine across the 64 chunks per (b, d) -> chunk-entry states.
__global__ __launch_bounds__(128, 1) void k_scan_pass2(
    const float* __restrict__ Aprod, const float* __restrict__ Bacc,
    float* __restrict__ Hst) {
  int tid = blockIdx.x * blockDim.x + threadIdx.x;     // [0, BATCH*D_INNER)
  int d = tid & (D_INNER - 1);
  int b = tid >> 9;
  float h[D_STATE];
#pragma unroll
  for (int s = 0; s < D_STATE; ++s) h[s] = 0.0f;
  for (int c = 0; c < NCHUNK; ++c) {
    size_t o = (((size_t)b * NCHUNK + c) * D_INNER + d) * D_STATE;
#pragma unroll
    for (int s = 0; s < D_STATE; ++s) {
      Hst[o + s] = h[s];
      h[s] = Aprod[o + s] * h[s] + Bacc[o + s];
    }
  }
}

// Pass 3: replay each chunk from its entry state; fuse y = sum(h*C) + x*D,
// SiLU(res) gating and bf16 conversion feeding out_proj.
__global__ __launch_bounds__(128, 1) void k_scan_pass3(
    const float* __restrict__ xdbl, const float* __restrict__ xcf,
    const float* __restrict__ xz,    // res = cols 512..1023
    const float* __restrict__ dtw, const float* __restrict__ dtb,
    const float* __restrict__ A_log, const float* __restrict__ Dp,
    const float* __restrict__ Hst, unsigned short* __restrict__ gbf) {
  int tid   = blockIdx.x * blockDim.x + threadIdx.x;
  int d     = tid & (D_INNER - 1);
  int chunk = (tid >> 9) & (NCHUNK - 1);
  int b     = tid >> 15;

  float wdt[DT_RANK], a[D_STATE], h[D_STATE];
#pragma unroll
  for (int r = 0; r < DT_RANK; ++r) wdt[r] = dtw[d * DT_RANK + r];
#pragma unroll
  for (int s = 0; s < D_STATE; ++s) {
    a[s] = -__expf(A_log[d * D_STATE + s]);
    h[s] = Hst[(size_t)tid * D_STATE + s];
  }
  float bias = dtb[d];
  float Dd   = Dp[d];

  size_t m0 = (size_t)b * SEQ + (size_t)chunk * CLEN;
  for (int l = 0; l < CLEN; ++l) {
    size_t m = m0 + l;
    const float* xr = xdbl + m * 48;
    __builtin_prefetch(xr + 48, 0, 0);   // warm next dt/B/C row
    float dtv = compute_dt(xr, wdt, bias);
    float xv  = xcf[m * D_INNER + d];
    float dtx = dtv * xv;
    float y = 0.0f;
#pragma unroll
    for (int s = 0; s < D_STATE; ++s) {
      float dA = __expf(dtv * a[s]);
      h[s] = dA * h[s] + dtx * xr[16 + s];
      y += h[s] * xr[32 + s];
    }
    y += xv * Dd;
    float rv = xz[m * (2 * D_INNER) + D_INNER + d];
    float g  = y * (rv * fast_sigmoid(rv));
    gbf[m * D_INNER + d] = f2bf(g);
  }
}

// ---------------- rank-1 scorer: scores = out . scorer_w + scorer_b ----------------
__global__ __launch_bounds__(256) void k_scorer(
    const float* __restrict__ outb, const float* __restrict__ sw,
    const float* __restrict__ sb, float* __restrict__ scores) {
  int row  = blockIdx.x * 8 + (threadIdx.x >> 5);     // 8 waves/block, one row each
  int lane = threadIdx.x & 31;
  float s = 0.0f;
#pragma unroll
  for (int k = lane; k < D_MODEL; k += 32)
    s += outb[(size_t)row * D_MODEL + k] * sw[k];
#pragma unroll
  for (int off = 16; off > 0; off >>= 1) s += __shfl_down(s, off, 32);
  if (lane == 0) scores[row] = s + sb[0];
}

extern "C" void kernel_launch(void* const* d_in, const int* in_sizes, int n_in,
                              void* d_out, int out_size, void* d_ws, size_t ws_size,
                              hipStream_t stream) {
  (void)in_sizes; (void)n_in; (void)out_size; (void)ws_size;
  const float* x      = (const float*)d_in[0];   // (4,4096,256)
  const float* w_in   = (const float*)d_in[1];   // (1024,256)
  const float* conv_w = (const float*)d_in[2];   // (512,1,4)
  const float* conv_b = (const float*)d_in[3];   // (512)
  const float* w_xp   = (const float*)d_in[4];   // (48,512)
  const float* dtw    = (const float*)d_in[5];   // (512,16)
  const float* dtb    = (const float*)d_in[6];   // (512)
  const float* A_log  = (const float*)d_in[7];   // (512,16)
  const float* Dp     = (const float*)d_in[8];   // (512)
  const float* w_out  = (const float*)d_in[9];   // (256,512)
  const float* sw     = (const float*)d_in[10];  // (1,256)
  const float* sb     = (const float*)d_in[11];  // (1)
  float* scores = (float*)d_out;                 // (4,4096,1) = 16384 f32

  // ---- workspace layout (256B aligned), with liveness-based reuse ----
  char* ws = (char*)d_ws;
  size_t off = 0;
  auto take = [&](size_t bytes) {
    size_t r = off; off += (bytes + 255) & ~(size_t)255; return r;
  };
  size_t o_xbf  = take((size_t)M_ROWS * D_MODEL * 2);             // x in bf16
  size_t o_win  = take((size_t)2 * D_INNER * D_MODEL * 2);        // in_proj_w bf16
  size_t o_xz   = take((size_t)M_ROWS * 2 * D_INNER * 4);         // xz f32; later reused as `out`
  size_t o_xcf  = take((size_t)M_ROWS * D_INNER * 4);             // conv+silu f32
  size_t o_xcbf = take((size_t)M_ROWS * D_INNER * 2);             // conv bf16; later reused as gated bf16
  size_t o_wxp  = take((size_t)48 * D_INNER * 2);                 // x_proj_w bf16
  size_t o_xdbl = take((size_t)M_ROWS * 48 * 4);                  // x_dbl f32
  size_t o_wout = take((size_t)D_MODEL * D_INNER * 2);            // out_proj_w bf16
  size_t o_ap   = take((size_t)BATCH * NCHUNK * D_INNER * D_STATE * 4);
  size_t o_bc   = take((size_t)BATCH * NCHUNK * D_INNER * D_STATE * 4);
  size_t o_hst  = take((size_t)BATCH * NCHUNK * D_INNER * D_STATE * 4);

  unsigned short* xbf    = (unsigned short*)(ws + o_xbf);
  unsigned short* winbf  = (unsigned short*)(ws + o_win);
  float*          xz     = (float*)(ws + o_xz);
  float*          outb   = (float*)(ws + o_xz);            // reuse: xz dead after scan
  float*          xcf    = (float*)(ws + o_xcf);
  unsigned short* xcbf   = (unsigned short*)(ws + o_xcbf);
  unsigned short* gbf    = (unsigned short*)(ws + o_xcbf); // reuse: xcbf dead after x_proj
  unsigned short* wxpbf  = (unsigned short*)(ws + o_wxp);
  float*          xdbl   = (float*)(ws + o_xdbl);
  unsigned short* woutbf = (unsigned short*)(ws + o_wout);
  float*          Ap     = (float*)(ws + o_ap);
  float*          Bc     = (float*)(ws + o_bc);
  float*          Hst    = (float*)(ws + o_hst);

  int n;
  n = M_ROWS * D_MODEL;
  k_f32_to_bf16<<<(n + 255) / 256, 256, 0, stream>>>(x, xbf, n);
  n = 2 * D_INNER * D_MODEL;
  k_f32_to_bf16<<<(n + 255) / 256, 256, 0, stream>>>(w_in, winbf, n);
  n = 48 * D_INNER;
  k_f32_to_bf16<<<(n + 255) / 256, 256, 0, stream>>>(w_xp, wxpbf, n);
  n = D_MODEL * D_INNER;
  k_f32_to_bf16<<<(n + 255) / 256, 256, 0, stream>>>(w_out, woutbf, n);

  // 64-row m-groups: M_ROWS/64 = 256 waves of work in M; 4 waves/block -> grid.x=64
  const int MGRP_BLOCKS = M_ROWS / 64 / 4;   // 64

  // in_proj: (16384 x 256) * (1024 x 256)^T -> xz (16384 x 1024); Ngroups = 16
  k_wmma_gemm_bf16<4, D_MODEL, D_MODEL, 2 * D_INNER>
      <<<dim3(MGRP_BLOCKS, 16), 128, 0, stream>>>(xbf, winbf, xz, D_MODEL);

  // causal depthwise conv + SiLU on xz[:, :512]
  n = M_ROWS * D_INNER;
  k_conv_silu<<<(n + 255) / 256, 256, 0, stream>>>(xz, conv_w, conv_b, xcf, xcbf);

  // x_proj: (16384 x 512) * (48 x 512)^T -> x_dbl (16384 x 48); one n-group of 3 tiles
  k_wmma_gemm_bf16<3, D_INNER, D_INNER, 48>
      <<<dim3(MGRP_BLOCKS, 1), 128, 0, stream>>>(xcbf, wxpbf, xdbl, D_INNER);

  // chunked selective scan (dt recomputed in-register; fused gate epilogue)
  k_scan_pass1<<<(BATCH * NCHUNK * D_INNER) / 128, 128, 0, stream>>>(
      xdbl, xcf, dtw, dtb, A_log, Ap, Bc);
  k_scan_pass2<<<(BATCH * D_INNER) / 128, 128, 0, stream>>>(Ap, Bc, Hst);
  k_scan_pass3<<<(BATCH * NCHUNK * D_INNER) / 128, 128, 0, stream>>>(
      xdbl, xcf, xz, dtw, dtb, A_log, Dp, Hst, gbf);

  // out_proj: (16384 x 512) * (256 x 512)^T -> out (16384 x 256); Ngroups = 4
  k_wmma_gemm_bf16<4, D_INNER, D_INNER, D_MODEL>
      <<<dim3(MGRP_BLOCKS, 4), 128, 0, stream>>>(gbf, woutbf, outb, D_INNER);

  // scorer
  k_scorer<<<M_ROWS / 8, 256, 0, stream>>>(outb, sw, sb, scores);
}
